// SlowFastSingleGlobalConHead_14018773254286
// MI455X (gfx1250) — matrix operand order repaired and checked by reading, not compiled, not run on hardware
//
#include <hip/hip_runtime.h>
#include <hip/hip_bf16.h>
#include <math.h>

typedef __attribute__((ext_vector_type(16))) _Float16 v16h;
typedef __attribute__((ext_vector_type(8)))  _Float16 v8h;
typedef __attribute__((ext_vector_type(8)))  float    v8f;

#define BATCH 64
#define DDIM  2304   // 2048 slow + 256 fast channels
#define HDIM  256    // head dim
#define INV_TEMP (1.0f / 0.07f)

// ---------------------------------------------------------------------------
// 1) Spatial mean pool: one wave per (b,c) slice; slice is contiguous, so
//    lane-strided float4 reads are fully coalesced. ~617MB streamed => the
//    bandwidth-bound bulk of this workload.
// ---------------------------------------------------------------------------
__global__ void pool_kernel(const float* __restrict__ src, int n4, float inv_spatial,
                            float* __restrict__ dst, int C, int colOff) {
    int wave = blockIdx.x * (blockDim.x >> 5) + (threadIdx.x >> 5);
    int lane = threadIdx.x & 31;
    const float4* p = reinterpret_cast<const float4*>(src) + (size_t)wave * n4;
    float s = 0.0f;
    for (int i = lane; i < n4; i += 32) {
        __builtin_prefetch(p + i + 128, 0, 1);   // global_prefetch_b8 (speculative, safe OOB)
        float4 v = p[i];
        s += v.x + v.y + v.z + v.w;
    }
    #pragma unroll
    for (int off = 16; off > 0; off >>= 1) s += __shfl_xor(s, off, 32);
    if (lane == 0) {
        int b = wave / C, c = wave - b * C;
        dst[(size_t)b * DDIM + colOff + c] = s * inv_spatial;
    }
}

// ---------------------------------------------------------------------------
// 2) W (2304x256 f32, row-major K x N) -> Wt (256x2304 f16, N-major) via LDS
//    tile transpose so both read and write stay coalesced.
// ---------------------------------------------------------------------------
__global__ void w_transpose_f16(const float* __restrict__ W, _Float16* __restrict__ Wt) {
    __shared__ float tile[32][33];
    int k0 = blockIdx.x * 32, n0 = blockIdx.y * 32;
    int tx = threadIdx.x, ty = threadIdx.y;
    tile[ty][tx] = W[(size_t)(k0 + ty) * HDIM + (n0 + tx)];
    __syncthreads();
    Wt[(size_t)(n0 + ty) * DDIM + (k0 + tx)] = (_Float16)tile[tx][ty];
}

// ---------------------------------------------------------------------------
// 3/5) Row L2 normalize (optional bias), emit f16 row for the WMMA GEMMs.
// ---------------------------------------------------------------------------
__global__ void norm_rows_f16(const float* __restrict__ src, const float* __restrict__ bias,
                              _Float16* __restrict__ dst, int N) {
    __shared__ float red[256];
    __shared__ float s_inv;
    int row = blockIdx.x;
    const float* r = src + (size_t)row * N;
    float ss = 0.0f;
    for (int j = threadIdx.x; j < N; j += blockDim.x) {
        float v = r[j] + (bias ? bias[j] : 0.0f);
        ss += v * v;
    }
    red[threadIdx.x] = ss; __syncthreads();
    for (int s2 = blockDim.x >> 1; s2 > 0; s2 >>= 1) {
        if (threadIdx.x < s2) red[threadIdx.x] += red[threadIdx.x + s2];
        __syncthreads();
    }
    if (threadIdx.x == 0) s_inv = 1.0f / fmaxf(sqrtf(red[0]), 1e-12f);
    __syncthreads();
    float inv = s_inv;
    for (int j = threadIdx.x; j < N; j += blockDim.x) {
        float v = r[j] + (bias ? bias[j] : 0.0f);
        dst[(size_t)row * N + j] = (_Float16)(v * inv);
    }
}

// ---------------------------------------------------------------------------
// WMMA fragment loaders matching the CDNA5 16-bit VGPR layouts (ISA 7.12.2).
// A (16x32):  lanes 0-15 hold M=lane, K chunks [0..7] and [16..23];
//             lanes 16-31 hold K chunks [8..15] and [24..31].
// B (32x16):  lanes 0-15 hold N=lane, K chunks [0..7] and [16..23];
//             lanes 16-31 hold K chunks [8..15] and [24..31].
// ---------------------------------------------------------------------------
__device__ __forceinline__ v16h load_a_frag(const _Float16* __restrict__ rowp, int kb, int lane) {
    int sh = (lane & 16) ? 8 : 0;
    v8h a0 = *reinterpret_cast<const v8h*>(rowp + kb + sh);
    v8h a1 = *reinterpret_cast<const v8h*>(rowp + kb + sh + 16);
    v16h a;
    #pragma unroll
    for (int i = 0; i < 8; ++i) { a[i] = a0[i]; a[8 + i] = a1[i]; }
    return a;
}

__device__ __forceinline__ v16h load_b_frag(const _Float16* __restrict__ rowp, int kb, int lane) {
    int sh = (lane & 16) ? 8 : 0;
    v8h b0 = *reinterpret_cast<const v8h*>(rowp + kb + sh);
    v8h b1 = *reinterpret_cast<const v8h*>(rowp + kb + sh + 16);
    v16h b;
    #pragma unroll
    for (int i = 0; i < 8; ++i) { b[i] = b0[i]; b[8 + i] = b1[i]; }
    return b;
}

// ---------------------------------------------------------------------------
// 4/6) Generic NT GEMM via v_wmma_f32_16x16x32_f16: Y = scale * (A @ Bt^T).
// A: (heads, M, K) f16 row-major.  Bt: (heads, N, K) f16 row-major.
// One wave per 16x16 output tile; full wave active (EXEC==all-1s).
// ---------------------------------------------------------------------------
__global__ void wmma_gemm_nt(const _Float16* __restrict__ A, const _Float16* __restrict__ Bt,
                             float* __restrict__ Y, int M, int N, int K,
                             size_t aHeadStride, size_t bHeadStride, size_t yHeadStride,
                             float scale) {
    int wave = blockIdx.x * (blockDim.x >> 5) + (threadIdx.x >> 5);
    int lane = threadIdx.x & 31;
    int ntiles = N >> 4;
    int tilesPerHead = (M >> 4) * ntiles;
    int head = wave / tilesPerHead;
    int t    = wave - head * tilesPerHead;
    int mT = t / ntiles, nT = t - mT * ntiles;

    const _Float16* Arow = A  + head * aHeadStride + (size_t)(mT * 16 + (lane & 15)) * K;
    const _Float16* Brow = Bt + head * bHeadStride + (size_t)(nT * 16 + (lane & 15)) * K;

    v8f c = {};
    #pragma unroll 4
    for (int kb = 0; kb < K; kb += 32) {
        v16h a = load_a_frag(Arow, kb, lane);
        v16h b = load_b_frag(Brow, kb, lane);
        c = __builtin_amdgcn_wmma_f32_16x16x32_f16(false, a, false, b, (short)0, c, false, false);
    }

    float* yh = Y + head * yHeadStride;
    int n = nT * 16 + (lane & 15);
    int mbase = mT * 16 + ((lane >> 4) << 3);
    #pragma unroll
    for (int r = 0; r < 8; ++r)
        yh[(size_t)(mbase + r) * N + n] = c[r] * scale;
}

// ---------------------------------------------------------------------------
// 7) Only the diagonal of sim_labels matters for the targets:
//    pos_i = 0.5*(dot(labels_v[i],labels_v_lt[i]) + dot(labels_n[i],labels_n_lt[i]))
// ---------------------------------------------------------------------------
__global__ void diag_kernel(const float* __restrict__ lv, const float* __restrict__ lvlt,
                            const float* __restrict__ ln, const float* __restrict__ lnlt,
                            float* __restrict__ diag) {
    int i = threadIdx.x;
    float s = 0.0f;
    const float* a = lv + i * 97;   const float* b = lvlt + i * 97;
    for (int j = 0; j < 97; ++j)  s += a[j] * b[j];
    const float* c = ln + i * 300;  const float* d = lnlt + i * 300;
    for (int j = 0; j < 300; ++j) s += c[j] * d[j];
    diag[i] = 0.5f * s;
}

// ---------------------------------------------------------------------------
// 8) Loss: per-row log-softmax over sim (64x64), target rule per same_idx,
//    mean over rows -> scalar.
// ---------------------------------------------------------------------------
__global__ void loss_kernel(const float* __restrict__ sim, const float* __restrict__ diag,
                            const int* __restrict__ same_idx, float* __restrict__ out) {
    __shared__ float red[64];
    int i = threadIdx.x;
    const float* row = sim + i * 64;
    float m = -3.4e38f;
    for (int j = 0; j < 64; ++j) m = fmaxf(m, row[j]);
    float se = 0.0f;
    for (int j = 0; j < 64; ++j) se += expf(row[j] - m);
    float logZ = m + logf(se);

    float pos = diag[i];
    int si = same_idx[i];
    float L;
    if (si == 2) {                          // eye target
        L = -(row[i] - logZ);
    } else if (si == 0) {                   // uniform target
        float s = 0.0f;
        for (int j = 0; j < 64; ++j) s += (row[j] - logZ);
        L = -s * (1.0f / 64.0f);
    } else {                                // soft target from pos_i
        float neg = (1.0f - pos) / 63.0f;
        float s = 0.0f;
        for (int j = 0; j < 64; ++j) {
            float tgt = (j == i) ? pos : neg;
            s += tgt * (row[j] - logZ);
        }
        L = -s;
    }
    red[i] = L; __syncthreads();
    for (int s2 = 32; s2 > 0; s2 >>= 1) {
        if (i < s2) red[i] += red[i + s2];
        __syncthreads();
    }
    if (i == 0) out[0] = red[0] * (1.0f / 64.0f);
}

// ---------------------------------------------------------------------------
extern "C" void kernel_launch(void* const* d_in, const int* in_sizes, int n_in,
                              void* d_out, int out_size, void* d_ws, size_t ws_size,
                              hipStream_t stream) {
    const float* fpv_slow = (const float*)d_in[0];
    const float* fpv_fast = (const float*)d_in[1];
    const float* tpv_slow = (const float*)d_in[2];
    const float* tpv_fast = (const float*)d_in[3];
    const float* labels_v    = (const float*)d_in[4];
    const float* labels_n    = (const float*)d_in[5];
    const float* labels_v_lt = (const float*)d_in[6];
    const float* labels_n_lt = (const float*)d_in[7];
    const int*   same_idx    = (const int*)d_in[8];
    const float* W_fpv = (const float*)d_in[9];
    const float* b_fpv = (const float*)d_in[10];
    const float* W_tpv = (const float*)d_in[11];
    const float* b_tpv = (const float*)d_in[12];
    float* out = (float*)d_out;

    // Workspace layout (256B aligned slabs), ~4.4 MB total.
    char* ws = (char*)d_ws;
    size_t off = 0;
    auto alloc = [&](size_t bytes) -> void* {
        off = (off + 255) & ~(size_t)255;
        void* p = ws + off;
        off += bytes;
        return p;
    };
    float*    pooled = (float*)   alloc(2u * BATCH * DDIM * sizeof(float));     // [head][64][2304]
    _Float16* xf16   = (_Float16*)alloc(2u * BATCH * DDIM * sizeof(_Float16));  // normalized rows
    _Float16* Wt     = (_Float16*)alloc(2u * HDIM * DDIM * sizeof(_Float16));   // [head][256][2304]
    float*    y      = (float*)   alloc(2u * BATCH * HDIM * sizeof(float));     // pre-norm proj
    _Float16* ft16   = (_Float16*)alloc(2u * BATCH * HDIM * sizeof(_Float16));  // f / t rows (f16)
    float*    sim    = (float*)   alloc((size_t)BATCH * BATCH * sizeof(float));
    float*    diag   = (float*)   alloc(BATCH * sizeof(float));

    float*    pooled_f = pooled;
    float*    pooled_t = pooled + (size_t)BATCH * DDIM;
    _Float16* xf16_f   = xf16;
    _Float16* xf16_t   = xf16 + (size_t)BATCH * DDIM;
    _Float16* Wt_f     = Wt;
    _Float16* Wt_t     = Wt + (size_t)HDIM * DDIM;
    float*    y_f      = y;
    float*    y_t      = y + (size_t)BATCH * HDIM;
    _Float16* f16_f    = ft16;
    _Float16* f16_t    = ft16 + (size_t)BATCH * HDIM;

    // 1) Mean pooling (the bandwidth-bound bulk). slow: 392 el -> 98 f4; fast: 1568 -> 392 f4.
    pool_kernel<<<(BATCH * 2048) / 8, 256, 0, stream>>>(fpv_slow, 98,  1.0f / 392.0f,  pooled_f, 2048, 0);
    pool_kernel<<<(BATCH * 256)  / 8, 256, 0, stream>>>(fpv_fast, 392, 1.0f / 1568.0f, pooled_f, 256,  2048);
    pool_kernel<<<(BATCH * 2048) / 8, 256, 0, stream>>>(tpv_slow, 98,  1.0f / 392.0f,  pooled_t, 2048, 0);
    pool_kernel<<<(BATCH * 256)  / 8, 256, 0, stream>>>(tpv_fast, 392, 1.0f / 1568.0f, pooled_t, 256,  2048);

    // 2) W -> f16, transposed to N-major for contiguous-K B fragments.
    w_transpose_f16<<<dim3(DDIM / 32, HDIM / 32), dim3(32, 32), 0, stream>>>(W_fpv, Wt_f);
    w_transpose_f16<<<dim3(DDIM / 32, HDIM / 32), dim3(32, 32), 0, stream>>>(W_tpv, Wt_t);

    // 3) L2-normalize pooled rows -> f16 A matrices.
    norm_rows_f16<<<BATCH, 256, 0, stream>>>(pooled_f, nullptr, xf16_f, DDIM);
    norm_rows_f16<<<BATCH, 256, 0, stream>>>(pooled_t, nullptr, xf16_t, DDIM);

    // 4) WMMA GEMM: y[head] = x[head] @ W[head]   (M=64, N=256, K=2304; 128 waves)
    wmma_gemm_nt<<<16, 256, 0, stream>>>(xf16, Wt, y, BATCH, HDIM, DDIM,
                                         (size_t)BATCH * DDIM, (size_t)HDIM * DDIM,
                                         (size_t)BATCH * HDIM, 1.0f);

    // 5) bias + L2-norm -> f (f16) and t (f16)
    norm_rows_f16<<<BATCH, 256, 0, stream>>>(y_f, b_fpv, f16_f, HDIM);
    norm_rows_f16<<<BATCH, 256, 0, stream>>>(y_t, b_tpv, f16_t, HDIM);

    // 6) WMMA GEMM: sim = (f @ t^T) / TEMP   (M=N=64, K=256; 16 waves)
    wmma_gemm_nt<<<2, 256, 0, stream>>>(f16_f, f16_t, sim, BATCH, BATCH, HDIM,
                                        0, 0, 0, INV_TEMP);

    // 7) diag(sim_labels); 8) log-softmax loss -> scalar
    diag_kernel<<<1, 64, 0, stream>>>(labels_v, labels_v_lt, labels_n, labels_n_lt, diag);
    loss_kernel<<<1, 64, 0, stream>>>(sim, diag, same_idx, out);
}
